// CausalSelfAttention_52261162058515
// MI455X (gfx1250) — matrix operand order repaired and checked by reading, not compiled, not run on hardware
//
#include <hip/hip_runtime.h>
#include <math.h>

// ---------------------------------------------------------------------------
// MI455X (gfx1250) causal self-attention.
// Matrix path: v_wmma_f32_16x16x32_f16 (f16 operands, f32 accumulate).
// Data movement: Tensor Data Mover (tensor_load_to_lds) double-buffered into
// LDS so operand bytes are fetched once per 128-wide block, not once per wave.
// Softmax / RoPE / norm stay f32.
// ---------------------------------------------------------------------------

typedef _Float16     v16h __attribute__((ext_vector_type(16)));
typedef _Float16     h8   __attribute__((ext_vector_type(8)));
typedef float        v8f  __attribute__((ext_vector_type(8)));
typedef unsigned int su4  __attribute__((ext_vector_type(4)));
typedef unsigned int su8  __attribute__((ext_vector_type(8)));

#define WMMA16(A, Bm, Cm) \
  __builtin_amdgcn_wmma_f32_16x16x32_f16(false, (A), false, (Bm), (short)0, (Cm), false, false)

constexpr int kB = 4, kT = 2048, kC = 768, kH = 6, kD = 128;
constexpr int kBT = kB * kT;          // 8192
constexpr int kQKV = 3 * kC;          // 2304

// ---------------------------------------------------------------------------
// TDM: async 2D f16 tile load, global -> LDS (CDNA5 ISA 08_async_tensor §7/§8).
// D# group0: count=1 | lds_addr | global_addr[56:0] | type=2.
// D# group1: data_size=2B; tensor dims set huge (tiling is exact, no OOB);
//            tile_w elems contiguous, tile_h rows, row stride in elems.
// ---------------------------------------------------------------------------
__device__ __forceinline__ unsigned lds_off(const void* p) {
  return (unsigned)(unsigned long long)p;  // addr[31:0] == LDS byte offset
}

__device__ __forceinline__ void tdm_load_2d_f16(unsigned lds_byte_off,
                                                const _Float16* gptr,
                                                unsigned tile_w, unsigned tile_h,
                                                unsigned stride_elems) {
  unsigned long long ga = (unsigned long long)gptr;
  su4 g0;
  g0[0] = 1u;                                                  // count = 1
  g0[1] = lds_byte_off;                                        // lds_addr
  g0[2] = (unsigned)ga;                                        // global_addr lo
  g0[3] = (unsigned)((ga >> 32) & 0x01FFFFFFu) | (2u << 30);   // addr hi | type=2
  su8 g1;
  g1[0] = 1u << 16;                  // workgroup_mask=0, data_size=2B, no flags
  g1[1] = 0u;                        // abar addr=0, tensor_dim0[15:0] (=0x40000000)
  g1[2] = 0x4000u;                   // tensor_dim0[31:16] | tensor_dim1[15:0]
  g1[3] = 0x4000u | (tile_w << 16);  // tensor_dim1[31:16] | tile_dim0
  g1[4] = tile_h;                    // tile_dim1 | tile_dim2=0
  g1[5] = stride_elems;              // tensor_dim0_stride[31:0]
  g1[6] = 0u;                        // stride hi | dim1_stride lo (unused, 2D)
  g1[7] = 0u;
  asm volatile("tensor_load_to_lds %0, %1" :: "s"(g0), "s"(g1) : "memory");
}

// ---------------------------------------------------------------------------
// elementwise f32 -> f16 cast
// ---------------------------------------------------------------------------
__global__ void cast_f32_f16(const float* __restrict__ s, _Float16* __restrict__ d, int n) {
  int i = blockIdx.x * blockDim.x + threadIdx.x;
  if (i < n) d[i] = (_Float16)s[i];
}

// ---------------------------------------------------------------------------
// A-operand (16x32 f16, contiguous k = 16h + s order, same as B loads) from a row-major [rows][ldk] buffer.
// lane<16: M=lane, halves 0..15 -> K=c0..c0+15
// lane>=16: same M, K offset +16.   (c0 = k0 + hi*16)
// ---------------------------------------------------------------------------
__device__ __forceinline__ v16h load_a_tile(const _Float16* __restrict__ base,
                                            int ldk, int row, int k0, int hi) {
  const _Float16* p = base + (size_t)row * ldk + k0 + hi * 16;
  h8 lo = *(const h8*)p;
  h8 hh = *(const h8*)(p + 8);
  v16h a;
#pragma unroll
  for (int z = 0; z < 8; ++z) { a[z] = lo[z]; a[z + 8] = hh[z]; }
  return a;
}

// ---------------------------------------------------------------------------
// GEMM: C[M,N](f32) = A[M,K](f16 rm) * W[N,K]^T (f16 rm).
// Block 256 = 8 waves -> 128x128 tile; TDM double-buffers 128x32 A/B panels
// in LDS; each wave computes a 32x64 sub-tile (2x4 WMMA accumulators).
// ---------------------------------------------------------------------------
__global__ __launch_bounds__(256) void gemm_f16_wmma(
    const _Float16* __restrict__ A, const _Float16* __restrict__ W,
    float* __restrict__ C, int M, int N, int K) {
  __shared__ __align__(64) _Float16 Apan[2][128 * 32];   // [row][k] 8KB each
  __shared__ __align__(64) _Float16 Bpan[2][128 * 32];   // [n][k]   8KB each

  const int tid = threadIdx.x;
  const int lane = tid & 31;
  const int wid = tid >> 5;
  const int hi = (lane >> 4) & 1;
  const int l16 = lane & 15;
  const int m0 = blockIdx.y * 128;
  const int n0 = blockIdx.x * 128;
  const int wm = (wid >> 1) * 32;   // 0,32,64,96
  const int wn = (wid & 1) * 64;    // 0,64
  const int ksteps = K >> 5;

  if (wid == 0) {
    tdm_load_2d_f16(lds_off(Apan[0]), A + (size_t)m0 * K, 32, 128, K);
    tdm_load_2d_f16(lds_off(Bpan[0]), W + (size_t)n0 * K, 32, 128, K);
  }

  v8f acc[2][4];
#pragma unroll
  for (int i = 0; i < 2; ++i)
#pragma unroll
    for (int j = 0; j < 4; ++j) acc[i][j] = (v8f)0.0f;

  for (int ks = 0; ks < ksteps; ++ks) {
    if (wid == 0) {
      if (ks + 1 < ksteps) {
        const int kb = (ks + 1) << 5;
        tdm_load_2d_f16(lds_off(Apan[(ks + 1) & 1]), A + (size_t)m0 * K + kb, 32, 128, K);
        tdm_load_2d_f16(lds_off(Bpan[(ks + 1) & 1]), W + (size_t)n0 * K + kb, 32, 128, K);
        __builtin_amdgcn_s_wait_tensorcnt(2);  // oldest pair (this step) done
      } else {
        __builtin_amdgcn_s_wait_tensorcnt(0);
      }
    }
    __syncthreads();  // panel (ks) visible to all waves

    const _Float16* Ap = Apan[ks & 1];
    const _Float16* Bp = Bpan[ks & 1];
    v16h a[2], b[4];
#pragma unroll
    for (int i = 0; i < 2; ++i)
      a[i] = load_a_tile(Ap, 32, wm + 16 * i + l16, 0, hi);
#pragma unroll
    for (int j = 0; j < 4; ++j)
      b[j] = *(const v16h*)(Bp + (size_t)(wn + 16 * j + l16) * 32 + hi * 16);
#pragma unroll
    for (int i = 0; i < 2; ++i)
#pragma unroll
      for (int j = 0; j < 4; ++j)
        acc[i][j] = WMMA16(a[i], b[j], acc[i][j]);

    __syncthreads();  // everyone done before buffer is re-filled
  }

#pragma unroll
  for (int i = 0; i < 2; ++i)
#pragma unroll
    for (int j = 0; j < 4; ++j)
#pragma unroll
      for (int r = 0; r < 8; ++r)
        C[(size_t)(m0 + wm + 16 * i + r + hi * 8) * N + n0 + wn + 16 * j + l16] =
            acc[i][j][r];
}

// ---------------------------------------------------------------------------
// Fused RoPE + norm + relayout (exact reference semantics: var/(D-1),
// 1/(sqrt(var)+eps)); Q gets 1/sqrt(D) folded in; V written transposed
// [BH, D, T] so the flash PV B-operand / TDM tile is contiguous.
// ---------------------------------------------------------------------------
__global__ __launch_bounds__(128) void rope_norm_relayout(
    const float* __restrict__ qkv, const float* __restrict__ cosT,
    const float* __restrict__ sinT, _Float16* __restrict__ Qh,
    _Float16* __restrict__ Kh, _Float16* __restrict__ Vt) {
  const int w = blockIdx.x * 4 + (threadIdx.x >> 5);
  const int lane = threadIdx.x & 31;
  const int row = w % 18;
  const int bt = w / 18;
  const int b = bt / kT, t = bt % kT;
  const int seg = row / 6, head = row % 6;
  const float* src = qkv + (size_t)bt * kQKV + seg * kC + head * kD;

  if (seg < 2) {
    float x1a = src[lane],      x1b = src[lane + 32];
    float x2a = src[lane + 64], x2b = src[lane + 96];
    float ca = cosT[t * 64 + lane], cb = cosT[t * 64 + lane + 32];
    float sa = sinT[t * 64 + lane], sb = sinT[t * 64 + lane + 32];
    float r0 =  x1a * ca + x2a * sa;
    float r1 =  x1b * cb + x2b * sb;
    float r2 = -x1a * sa + x2a * ca;
    float r3 = -x1b * sb + x2b * cb;
    float sum = r0 + r1 + r2 + r3;
#pragma unroll
    for (int m = 1; m < 32; m <<= 1) sum += __shfl_xor(sum, m, 32);
    float mean = sum * (1.0f / 128.0f);
    float d0 = r0 - mean, d1 = r1 - mean, d2 = r2 - mean, d3 = r3 - mean;
    float sq = d0 * d0 + d1 * d1 + d2 * d2 + d3 * d3;
#pragma unroll
    for (int m = 1; m < 32; m <<= 1) sq += __shfl_xor(sq, m, 32);
    float inv = 1.0f / (sqrtf(sq * (1.0f / 127.0f)) + 1e-6f);
    if (seg == 0) inv *= 0.08838834764831845f;  // fold 1/sqrt(128)
    _Float16* dst = (seg == 0 ? Qh : Kh) + ((size_t)(b * kH + head) * kT + t) * kD;
    dst[lane]      = (_Float16)(d0 * inv);
    dst[lane + 32] = (_Float16)(d1 * inv);
    dst[lane + 64] = (_Float16)(d2 * inv);
    dst[lane + 96] = (_Float16)(d3 * inv);
  } else {
    _Float16* dst = Vt + (size_t)(b * kH + head) * kD * kT;
    dst[(size_t)(lane)      * kT + t] = (_Float16)src[lane];
    dst[(size_t)(lane + 32) * kT + t] = (_Float16)src[lane + 32];
    dst[(size_t)(lane + 64) * kT + t] = (_Float16)src[lane + 64];
    dst[(size_t)(lane + 96) * kT + t] = (_Float16)src[lane + 96];
  }
}

// ---------------------------------------------------------------------------
// Flash attention: 4 waves / block share TDM-staged K,V tiles (double
// buffered); each wave owns a 16-row query tile (64 rows per block).
// Key loop is block-uniform over [0, t0b+64); frontier tiles are masked,
// fully-masked tiles degenerate to alpha=1 / rs=0 in the online softmax.
// ---------------------------------------------------------------------------
__global__ __launch_bounds__(128) void flash_attn_wmma(
    const _Float16* __restrict__ Qh,  // [BH, T, D], pre-scaled by 1/sqrt(D)
    const _Float16* __restrict__ Kh,  // [BH, T, D]
    const _Float16* __restrict__ Vt,  // [BH, D, T]
    _Float16* __restrict__ Y) {       // [B, T, H*D]
  __shared__ __align__(64) _Float16 Kpan[2][32 * 128];  // [key][d]  8KB each
  __shared__ __align__(64) _Float16 Vpan[2][128 * 32];  // [d][key]  8KB each
  __shared__ float Plds[4][16 * 33];                    // per-wave P transpose

  const int tid = threadIdx.x;
  const int lane = tid & 31;
  const int wid = tid >> 5;
  const int hi = (lane >> 4) & 1;
  const int l16 = lane & 15;
  const int t0b = blockIdx.x * 64;
  const int t0 = t0b + wid * 16;
  const int bh = blockIdx.y;
  const int b = bh / kH, h = bh % kH;

  const _Float16* Qb = Qh + (size_t)bh * kT * kD;
  const _Float16* Kb = Kh + (size_t)bh * kT * kD;
  const _Float16* Vb = Vt + (size_t)bh * kD * kT;
  float* pw = Plds[wid];

  v16h qa[4];
#pragma unroll
  for (int dd = 0; dd < 4; ++dd)
    qa[dd] = load_a_tile(Qb, kD, t0 + l16, dd * 32, hi);

  v8f o[8];
#pragma unroll
  for (int j = 0; j < 8; ++j) o[j] = (v8f)0.0f;
  float mrow[8], lrow[8];
#pragma unroll
  for (int r = 0; r < 8; ++r) { mrow[r] = -1e30f; lrow[r] = 0.0f; }

  const int nTiles = (t0b + 64) >> 5;
  if (wid == 0) {
    tdm_load_2d_f16(lds_off(Kpan[0]), Kb, 128, 32, kD);        // [32][128]
    tdm_load_2d_f16(lds_off(Vpan[0]), Vb, 32, 128, kT);        // [128][32]
  }

  for (int it = 0; it < nTiles; ++it) {
    const int s0 = it * 32;
    if (wid == 0) {
      if (it + 1 < nTiles) {
        const int s1 = s0 + 32;
        tdm_load_2d_f16(lds_off(Kpan[(it + 1) & 1]), Kb + (size_t)s1 * kD, 128, 32, kD);
        tdm_load_2d_f16(lds_off(Vpan[(it + 1) & 1]), Vb + s1, 32, 128, kT);
        __builtin_amdgcn_s_wait_tensorcnt(2);
      } else {
        __builtin_amdgcn_s_wait_tensorcnt(0);
      }
    }
    __syncthreads();

    const _Float16* Kp = Kpan[it & 1];
    const _Float16* Vp = Vpan[it & 1];

    // S = Q*K^T : two 16x16 tiles from LDS K panel
    v8f s[2];
    s[0] = (v8f)0.0f; s[1] = (v8f)0.0f;
#pragma unroll
    for (int half = 0; half < 2; ++half)
#pragma unroll
      for (int dd = 0; dd < 4; ++dd) {
        v16h kb = *(const v16h*)(Kp + (size_t)(16 * half + l16) * kD + dd * 32 + hi * 16);
        s[half] = WMMA16(qa[dd], kb, s[half]);
      }

    if (s0 + 31 > t0) {  // causal frontier (per-wave, uniform in wave)
#pragma unroll
      for (int half = 0; half < 2; ++half)
#pragma unroll
        for (int r = 0; r < 8; ++r) {
          int rowa = t0 + r + hi * 8;
          int key = s0 + 16 * half + l16;
          if (key > rowa) s[half][r] = -1e30f;
        }
    }

    // online softmax, rows live across 16-lane halves (C-matrix layout)
    float alpha[8];
#pragma unroll
    for (int r = 0; r < 8; ++r) {
      float v = fmaxf(s[0][r], s[1][r]);
#pragma unroll
      for (int m = 1; m < 16; m <<= 1) v = fmaxf(v, __shfl_xor(v, m, 32));
      float mnew = fmaxf(mrow[r], v);
      alpha[r] = __expf(mrow[r] - mnew);
      mrow[r] = mnew;
      s[0][r] = __expf(s[0][r] - mnew);
      s[1][r] = __expf(s[1][r] - mnew);
      float rs = s[0][r] + s[1][r];
#pragma unroll
      for (int m = 1; m < 16; m <<= 1) rs += __shfl_xor(rs, m, 32);
      lrow[r] = lrow[r] * alpha[r] + rs;
    }
#pragma unroll
    for (int j = 0; j < 8; ++j)
#pragma unroll
      for (int r = 0; r < 8; ++r) o[j][r] *= alpha[r];

    // P: C-layout f32 -> A-layout f16 via per-wave LDS slice (in-order DS)
    asm volatile("s_wait_dscnt 0" ::: "memory");
#pragma unroll
    for (int half = 0; half < 2; ++half)
#pragma unroll
      for (int r = 0; r < 8; ++r)
        pw[(r + hi * 8) * 33 + 16 * half + l16] = s[half][r];
    asm volatile("s_wait_dscnt 0" ::: "memory");
    v16h pa;
#pragma unroll
    for (int z = 0; z < 16; ++z) {
      int c = z + hi * 16;
      pa[z] = (_Float16)pw[l16 * 33 + c];
    }

    // O += P*V from LDS V^T panel
#pragma unroll
    for (int j = 0; j < 8; ++j) {
      v16h vb = *(const v16h*)(Vp + (size_t)(j * 16 + l16) * 32 + hi * 16);
      o[j] = WMMA16(pa, vb, o[j]);
    }

    __syncthreads();  // all waves done before panel re-fill
  }

#pragma unroll
  for (int r = 0; r < 8; ++r) {
    float inv = 1.0f / lrow[r];
    size_t rowoff = ((size_t)b * kT + t0 + r + hi * 8) * (kH * kD) + h * kD;
#pragma unroll
    for (int j = 0; j < 8; ++j)
      Y[rowoff + j * 16 + l16] = (_Float16)(o[j][r] * inv);
  }
}

// ---------------------------------------------------------------------------
// host-side launcher
// ---------------------------------------------------------------------------
extern "C" void kernel_launch(void* const* d_in, const int* in_sizes, int n_in,
                              void* d_out, int out_size, void* d_ws, size_t ws_size,
                              hipStream_t stream) {
  (void)in_sizes; (void)n_in; (void)out_size; (void)ws_size;
  const float* x    = (const float*)d_in[0];
  const float* cosT = (const float*)d_in[1];
  const float* sinT = (const float*)d_in[2];
  const float* wq   = (const float*)d_in[3];
  const float* wk   = (const float*)d_in[4];
  const float* wv   = (const float*)d_in[5];
  const float* wp   = (const float*)d_in[6];

  char* ws = (char*)d_ws;
  size_t off = 0;
  auto carve = [&](size_t bytes) -> void* {
    void* p = ws + off;
    off = (off + bytes + 255) & ~(size_t)255;
    return p;
  };
  _Float16* xh    = (_Float16*)carve((size_t)kBT * kC * 2);
  _Float16* wqkvh = (_Float16*)carve((size_t)kQKV * kC * 2);
  _Float16* wph   = (_Float16*)carve((size_t)kC * kC * 2);
  float*    qkv   = (float*)   carve((size_t)kBT * kQKV * 4);
  _Float16* Qh    = (_Float16*)carve((size_t)kBT * kC * 2);
  _Float16* Kh    = (_Float16*)carve((size_t)kBT * kC * 2);
  _Float16* Vt    = (_Float16*)carve((size_t)kBT * kC * 2);
  _Float16* Yh    = (_Float16*)carve((size_t)kBT * kC * 2);

  const int nX = kBT * kC;
  const int nW = kC * kC;
  cast_f32_f16<<<(nX + 255) / 256, 256, 0, stream>>>(x, xh, nX);
  cast_f32_f16<<<(nW + 255) / 256, 256, 0, stream>>>(wq, wqkvh,          nW);
  cast_f32_f16<<<(nW + 255) / 256, 256, 0, stream>>>(wk, wqkvh +   nW,   nW);
  cast_f32_f16<<<(nW + 255) / 256, 256, 0, stream>>>(wv, wqkvh + 2 * nW, nW);
  cast_f32_f16<<<(nW + 255) / 256, 256, 0, stream>>>(wp, wph,            nW);

  // qkv = x * [wq;wk;wv]^T : [8192,768] x [768,2304], 128x128 tiles
  gemm_f16_wmma<<<dim3(kQKV / 128, kBT / 128), 256, 0, stream>>>(
      xh, wqkvh, qkv, kBT, kQKV, kC);

  rope_norm_relayout<<<(kBT * 18) / 4, 128, 0, stream>>>(
      qkv, cosT, sinT, Qh, Kh, Vt);

  // flash: (T/64) x (B*H) blocks of 4 waves
  flash_attn_wmma<<<dim3(kT / 64, kB * kH), 128, 0, stream>>>(Qh, Kh, Vt, Yh);

  // out = y * wproj^T : [8192,768] x [768,768]
  gemm_f16_wmma<<<dim3(kC / 128, kBT / 128), 256, 0, stream>>>(
      Yh, wph, (float*)d_out, kBT, kC, kC);
}
